// ActorNetSpikingwithRSTDP_77678778515916
// MI455X (gfx1250) — matrix operand order repaired and checked by reading, not compiled, not run on hardware
//
#include <hip/hip_runtime.h>

// ---------------------------------------------------------------------------
// SNN forward (4-layer LIF, T=50) for MI455X (gfx1250, wave32, WMMA).
// B=4096, S=256, H=256, A=2, T=50.
// bf16 WMMA 16x16x32, fp32 accumulation. LIF state in VGPRs (s recomputed
// from v). Spikes cross waves via LDS. x gathered as aligned float2
// (2 timesteps per load). Register budget kept < 256 to avoid spills:
// BTILE=32 (2 m-tiles/wave), layer-4 state spread over waves 0-1.
// ---------------------------------------------------------------------------

#define B_TOT   4096
#define S_DIM   256
#define H_DIM   256
#define T_STEPS 50
#define A_DIM   2

#define BTILE   32      // batch rows per workgroup
#define VTH     0.5f
#define CDECAY  0.5f
#define VDECAY  0.75f

typedef __attribute__((ext_vector_type(16))) __bf16 v16bf;
typedef __attribute__((ext_vector_type(8)))  __bf16 v8bf;
typedef __attribute__((ext_vector_type(8)))  float  v8f;

// --- WMMA A fragment (16x32 bf16) from row-major [row][256] bf16 buffer. ---
// lanes 0-15: M=lane, K chunks {k0..k0+7, k0+16..k0+23};
// lanes 16-31: M=lane-16, chunks {k0+8..k0+15, k0+24..k0+31}.
__device__ __forceinline__ v16bf load_A(const __bf16* buf, int mbase, int lane, int k0) {
  const int row = mbase + (lane & 15);
  const int h   = (lane >> 4) * 8;
  const v8bf lo = *(const v8bf*)(buf + row * 256 + k0 + h);
  const v8bf hi = *(const v8bf*)(buf + row * 256 + k0 + 16 + h);
  v16bf a;
#pragma unroll
  for (int i = 0; i < 8; ++i) { a[i] = lo[i]; a[8 + i] = hi[i]; }
  return a;
}

// --- WMMA B fragment (32x16 bf16) from row-major transposed weights Wt[N][256].
// lanes 0-15: N=lane, K chunks {k0..k0+7, k0+16..k0+23};
// lanes 16-31: N=lane-16, chunks {k0+8..k0+15, k0+24..k0+31}.
__device__ __forceinline__ v16bf load_B(const __bf16* wt, int n0, int lane, int k0) {
  const int n  = n0 + (lane & 15);
  const int kb = k0 + ((lane >> 4) << 3);
  const v8bf lo = *(const v8bf*)(wt + n * 256 + kb);
  const v8bf hi = *(const v8bf*)(wt + n * 256 + kb + 16);
  v16bf b;
#pragma unroll
  for (int i = 0; i < 8; ++i) { b[i] = lo[i]; b[8 + i] = hi[i]; }
  return b;
}

// --- One LIF layer, this wave's 16 columns (n0..n0+15), 2 m-tiles.
// cur = 0.5*u + pre@W + b ; volt = (v>VTH ? 0 : 0.75*v) + cur ; spike = volt>VTH
// Software-pipelined: next k-step's A/B fragments are loaded before the WMMAs.
__device__ __forceinline__ void lif_layer(const __bf16* inbuf, const __bf16* wt,
                                          int n0, int lane, float bias,
                                          v8f* u, v8f* v, __bf16* outbuf) {
  v8f acc[2];
#pragma unroll
  for (int m = 0; m < 2; ++m)
#pragma unroll
    for (int e = 0; e < 8; ++e) acc[m][e] = CDECAY * u[m][e] + bias;

  v16bf bf = load_B(wt, n0, lane, 0);
  v16bf a0 = load_A(inbuf, 0, lane, 0);
  v16bf a1 = load_A(inbuf, 16, lane, 0);
#pragma unroll
  for (int ks = 0; ks < 8; ++ks) {
    v16bf nbf, na0, na1;
    if (ks < 7) {
      nbf = load_B(wt, n0, lane, (ks + 1) * 32);
      na0 = load_A(inbuf, 0, lane, (ks + 1) * 32);
      na1 = load_A(inbuf, 16, lane, (ks + 1) * 32);
    }
    acc[0] = __builtin_amdgcn_wmma_f32_16x16x32_bf16(
        false, a0, false, bf, (short)0, acc[0], false, false);
    acc[1] = __builtin_amdgcn_wmma_f32_16x16x32_bf16(
        false, a1, false, bf, (short)0, acc[1], false, false);
    bf = nbf; a0 = na0; a1 = na1;
  }

  const int col   = n0 + (lane & 15);
  const int rbase = 8 * (lane >> 4);
#pragma unroll
  for (int m = 0; m < 2; ++m) {
    u[m] = acc[m];
#pragma unroll
    for (int e = 0; e < 8; ++e) {
      const float vold = v[m][e];
      const float volt = (vold > VTH ? 0.0f : vold * VDECAY) + acc[m][e];
      v[m][e] = volt;
      outbuf[(16 * m + rbase + e) * 256 + col] =
          (volt > VTH) ? (__bf16)1.0f : (__bf16)0.0f;
    }
  }
}

__global__ __launch_bounds__(512) void snn_forward(
    const float* __restrict__ x,
    const float* __restrict__ b1, const float* __restrict__ b2,
    const float* __restrict__ b3, const float* __restrict__ b4,
    const __bf16* __restrict__ wt1, const __bf16* __restrict__ wt2,
    const __bf16* __restrict__ wt3, const __bf16* __restrict__ wt4,
    float* __restrict__ out) {
  __shared__ __bf16 xbuf[2][BTILE * 256];  // 32 KB: two timesteps of input
  __shared__ __bf16 sA[BTILE * 256];       // 16 KB
  __shared__ __bf16 sB[BTILE * 256];       // 16 KB

  const int tid   = threadIdx.x;
  const int lane  = tid & 31;
  const int wave  = tid >> 5;
  const int bbase = blockIdx.x * BTILE;
  const int n0    = wave * 16;

  v8f u1[2], v1[2], u2[2], v2[2], u3[2], v3[2];
  v8f u4, v4, sum4;  // waves 0 and 1: one 16x16 layer-4 m-tile each
#pragma unroll
  for (int m = 0; m < 2; ++m)
#pragma unroll
    for (int e = 0; e < 8; ++e) {
      u1[m][e] = 0.f; v1[m][e] = 0.f; u2[m][e] = 0.f; v2[m][e] = 0.f;
      u3[m][e] = 0.f; v3[m][e] = 0.f;
    }
#pragma unroll
  for (int e = 0; e < 8; ++e) { u4[e] = 0.f; v4[e] = 0.f; sum4[e] = 0.f; }

  const int nn = lane & 15;
  const float bias1 = b1[n0 + nn];
  const float bias2 = b2[n0 + nn];
  const float bias3 = b3[n0 + nn];
  const float bias4 = (nn < A_DIM) ? b4[nn] : 0.0f;

#pragma unroll 1
  for (int t = 0; t < T_STEPS; t += 2) {
    // Cooperative gather of x[btile, :, t..t+1]; index (b*256+s)*50+t is even
    // when t is even -> 8-byte-aligned float2 loads.
#pragma unroll 4
    for (int i = 0; i < (BTILE * 256) / 512; ++i) {
      const int p = tid + i * 512;
      const int b = p >> 8;
      const int s = p & 255;
      const float* src = x + (size_t)((bbase + b) * 256 + s) * T_STEPS + t;
      const float2 xv = *(const float2*)src;
      if (t + 2 < T_STEPS) __builtin_prefetch(src + 2, 0, 0);
      xbuf[0][b * 256 + s] = (__bf16)xv.x;
      xbuf[1][b * 256 + s] = (__bf16)xv.y;
    }
    __syncthreads();

#pragma unroll 1
    for (int tt = 0; tt < 2; ++tt) {
      lif_layer(xbuf[tt], wt1, n0, lane, bias1, u1, v1, sA);
      __syncthreads();
      lif_layer(sA, wt2, n0, lane, bias2, u2, v2, sB);
      __syncthreads();
      lif_layer(sB, wt3, n0, lane, bias3, u3, v3, sA);  // s3 -> sA
      __syncthreads();

      if (wave < 2) {  // layer 4 (A=2 padded to 16 cols); wave w does m-tile w
        v8f acc;
#pragma unroll
        for (int e = 0; e < 8; ++e) acc[e] = CDECAY * u4[e] + bias4;

        v16bf bf = load_B(wt4, 0, lane, 0);
        v16bf af = load_A(sA, 16 * wave, lane, 0);
#pragma unroll
        for (int ks = 0; ks < 8; ++ks) {
          v16bf nbf, naf;
          if (ks < 7) {
            nbf = load_B(wt4, 0, lane, (ks + 1) * 32);
            naf = load_A(sA, 16 * wave, lane, (ks + 1) * 32);
          }
          acc = __builtin_amdgcn_wmma_f32_16x16x32_bf16(
              false, af, false, bf, (short)0, acc, false, false);
          bf = nbf; af = naf;
        }
        u4 = acc;
#pragma unroll
        for (int e = 0; e < 8; ++e) {
          const float vold = v4[e];
          const float volt = (vold > VTH ? 0.0f : vold * VDECAY) + acc[e];
          v4[e] = volt;
          sum4[e] += (volt > VTH) ? 1.0f : 0.0f;
        }
      }
      __syncthreads();  // protect sA (read by L4) before next step's L1 write
    }
  }

  if (wave < 2) {
    const float scale = 1.0f / ((float)T_STEPS * (float)T_STEPS);
    const int N = lane & 15;
    if (N < A_DIM) {
      const int rbase = 8 * (lane >> 4);
#pragma unroll
      for (int e = 0; e < 8; ++e)
        out[(bbase + 16 * wave + rbase + e) * A_DIM + N] = sum4[e] * scale;
    }
  }
}

// --- Weight prep: fp32 [K][N] -> bf16 transposed [N][K]; w4 zero-padded to 16.
__global__ void prep_weights(const float* __restrict__ w1,
                             const float* __restrict__ w2,
                             const float* __restrict__ w3,
                             const float* __restrict__ w4,
                             __bf16* __restrict__ wt1, __bf16* __restrict__ wt2,
                             __bf16* __restrict__ wt3, __bf16* __restrict__ wt4) {
  const int blk = blockIdx.x;
  const int k   = threadIdx.x;  // 0..255
  if (blk < 3 * 256) {
    const int layer = blk >> 8;
    const int n     = blk & 255;
    const float* w  = (layer == 0) ? w1 : ((layer == 1) ? w2 : w3);
    __bf16*      wt = (layer == 0) ? wt1 : ((layer == 1) ? wt2 : wt3);
    wt[n * 256 + k] = (__bf16)w[k * 256 + n];
  } else {
    const int n = blk - 3 * 256;  // 0..15
    wt4[n * 256 + k] = (n < A_DIM) ? (__bf16)w4[k * A_DIM + n] : (__bf16)0.0f;
  }
}

extern "C" void kernel_launch(void* const* d_in, const int* in_sizes, int n_in,
                              void* d_out, int out_size, void* d_ws, size_t ws_size,
                              hipStream_t stream) {
  (void)in_sizes; (void)n_in; (void)out_size; (void)ws_size;
  const float* x  = (const float*)d_in[0];
  const float* w1 = (const float*)d_in[1];
  const float* b1 = (const float*)d_in[2];
  const float* w2 = (const float*)d_in[3];
  const float* b2 = (const float*)d_in[4];
  const float* w3 = (const float*)d_in[5];
  const float* b3 = (const float*)d_in[6];
  const float* w4 = (const float*)d_in[7];
  const float* b4 = (const float*)d_in[8];

  __bf16* wt1 = (__bf16*)d_ws;
  __bf16* wt2 = wt1 + 256 * 256;
  __bf16* wt3 = wt2 + 256 * 256;
  __bf16* wt4 = wt3 + 256 * 256;  // 16*256 elems; total ws use ~401 KB

  prep_weights<<<3 * 256 + 16, 256, 0, stream>>>(w1, w2, w3, w4, wt1, wt2, wt3, wt4);
  snn_forward<<<B_TOT / BTILE, 512, 0, stream>>>(x, b1, b2, b3, b4,
                                                 wt1, wt2, wt3, wt4,
                                                 (float*)d_out);
}
